// CRT_15298673508395
// MI455X (gfx1250) — matrix operand order repaired and checked by reading, not compiled, not run on hardware
//
#include <hip/hip_runtime.h>
#include <hip/hip_bf16.h>

typedef __attribute__((ext_vector_type(16))) _Float16 v16h;
typedef __attribute__((ext_vector_type(8)))  float    v8f;

#define BATCH 8
static constexpr float BN_INV = 0.99999500003749981f;  // 1/sqrt(1+1e-5)

enum { MODE_BIAS = 0, MODE_RELU = 1, MODE_RES = 3, MODE_ACC = 4 };

union H2Pack { _Float16 h[2]; unsigned u; };

// ---------------------------------------------------------------------------
// WMMA fragment loaders (CDNA5 16x16x32 f16 layouts, ISA 7.12.2)
// A 16x32 f16: lanes 0-15 = rows M; vgpr p holds K pair:
//   K = 16*(p/4) + 8*(lane/16) + 2*(p%4) + s
// B 32x16 f16: lanes = cols N; vgpr p holds K pair: K = 16*(p/4) + 8*(lane/16) + 2*(p%4) + s
//   -> for lane (li,hh) the 16 B-values are K = 8*hh + 0..7 and 16 + 8*hh + 0..7 (as for A)
// D 16x16 f32: lane holds col N = lane%16, vgpr r = row M = r + 8*(lane/16)
// ---------------------------------------------------------------------------
__device__ __forceinline__ v16h load_a_frag_h(const _Float16* __restrict__ W, int ld,
                                              int m0, int kc, int li, int hh) {
  v16h a;
  const _Float16* row = W + (size_t)(m0 + li) * ld + kc;
#pragma unroll
  for (int p = 0; p < 8; ++p) {
    int kk = ((p >> 2) << 4) + (hh << 3) + ((p & 3) << 1);
    a[2 * p]     = row[kk];
    a[2 * p + 1] = row[kk + 1];
  }
  return a;
}

// B operand from LDS stored transposed [n][k] (f16, padded row stride):
// lane reads halves k = 8*hh + 0..7 and 16 + 8*hh + 0..7 -> 2x ds_load_b128.
__device__ __forceinline__ v16h load_b_lds_h(const _Float16* __restrict__ S, int stride,
                                             int kc, int li, int hh) {
  v16h b;
  const _Float16* src = S + li * stride + kc + (hh << 3);
#pragma unroll
  for (int i = 0; i < 16; ++i) b[i] = src[i + ((i >> 3) << 3)];
  return b;
}

// ---------------------------------------------------------------------------
// f32 -> f16 weight pre-conversion (run once per launch)
// ---------------------------------------------------------------------------
__global__ void cvt_f16_kernel(const float* __restrict__ in, _Float16* __restrict__ out,
                               int n) {
  int g = blockIdx.x * blockDim.x + threadIdx.x;
  if (g < n) out[g] = (_Float16)in[g];
}

// ---------------------------------------------------------------------------
// 1x1-conv GEMM: Y[b,m,n] = epilogue( sum_c W[m,c]*X[b,c,n] + bias[m] )
// X may be the channel-concat of X1 (C1 channels) and X2 (K-C1 channels).
// Block = 128 threads (4 waves) computes a 64x64 tile; W/X K-chunks staged
// through LDS as packed f16 with two-phase (load-all, then cvt+store)
// software pipelining; X stored transposed [n][k] (stride 40) so B fragments
// are contiguous b128 LDS loads. grid = (Np/64, M/64, B).
// ---------------------------------------------------------------------------
__global__ void __launch_bounds__(128)
gemm_conv(const float* __restrict__ W, const float* __restrict__ bias,
          const float* __restrict__ X1, const float* __restrict__ X2, int C1,
          float* __restrict__ Y, const float* __restrict__ R,
          int M, int K, int Np, int mode) {
  __shared__ __align__(16) _Float16 s_w[64 * 32];  // [m][k]          4 KB
  __shared__ __align__(16) _Float16 s_x[64 * 40];  // [n][k pad 40]   5 KB
  int bn0 = blockIdx.x * 64, bm0 = blockIdx.y * 64, b = blockIdx.z;
  int tid = threadIdx.x;
  int wave = tid >> 5, lane = tid & 31, li = lane & 15, hh = lane >> 4;
  const float* x1 = X1 + (size_t)b * C1 * Np;
  const float* x2 = X2 ? X2 + (size_t)b * (K - C1) * Np : nullptr;
  v8f acc[4] = {{}, {}, {}, {}};
  for (int kc = 0; kc < K; kc += 32) {
    // ---- phase 1: issue all staging loads (pipelined, no intermediate waits)
    float2 wv[8];
#pragma unroll
    for (int i = 0; i < 8; ++i) {
      int e = tid + i * 128;                 // W pair e = m*16 + k/2
      int m = e >> 4, kp = (e & 15) << 1;
      wv[i] = *(const float2*)&W[(size_t)(bm0 + m) * K + kc + kp];
    }
    float xv0[8], xv1[8];
#pragma unroll
    for (int i = 0; i < 8; ++i) {
      int e = tid + i * 128;                 // X pair (kp,n)
      int kp = e >> 6, n = e & 63;
      int c0 = kc + (kp << 1), c1 = c0 + 1;
      xv0[i] = (c0 < C1) ? x1[(size_t)c0 * Np + bn0 + n]
                         : x2[(size_t)(c0 - C1) * Np + bn0 + n];
      xv1[i] = (c1 < C1) ? x1[(size_t)c1 * Np + bn0 + n]
                         : x2[(size_t)(c1 - C1) * Np + bn0 + n];
    }
    // prefetch next K-chunk while this one is converted/consumed
    if (kc + 32 < K) {
      int m = tid >> 1, half = tid & 1;
      __builtin_prefetch(&W[(size_t)(bm0 + m) * K + kc + 32 + half * 16], 0, 3);
      int kk = tid >> 2, q = tid & 3;
      int c = kc + 32 + kk;
      const float* xp = (c < C1) ? &x1[(size_t)c * Np + bn0 + q * 16]
                                 : &x2[(size_t)(c - C1) * Np + bn0 + q * 16];
      __builtin_prefetch(xp, 0, 3);
    }
    // ---- phase 2: convert + LDS store
#pragma unroll
    for (int i = 0; i < 8; ++i) {
      int e = tid + i * 128;
      H2Pack pk;
      pk.h[0] = (_Float16)wv[i].x;
      pk.h[1] = (_Float16)wv[i].y;
      ((unsigned*)s_w)[e] = pk.u;
    }
#pragma unroll
    for (int i = 0; i < 8; ++i) {
      int e = tid + i * 128;
      int kp = e >> 6, n = e & 63;
      H2Pack pk;
      pk.h[0] = (_Float16)xv0[i];
      pk.h[1] = (_Float16)xv1[i];
      ((unsigned*)s_x)[n * 20 + kp] = pk.u;
    }
    __syncthreads();
    // A fragment for this wave's 16-row strip (reused across 4 N-subtiles)
    v16h a;
#pragma unroll
    for (int p = 0; p < 8; ++p) {
      int kk = ((p >> 2) << 4) + (hh << 3) + ((p & 3) << 1);
      const _Float16* rw = &s_w[(wave * 16 + li) * 32 + kk];
      a[2 * p] = rw[0];
      a[2 * p + 1] = rw[1];
    }
#pragma unroll
    for (int j = 0; j < 4; ++j) {
      v16h bf = load_b_lds_h(s_x, 40, 0, j * 16 + li, hh);
      acc[j] = __builtin_amdgcn_wmma_f32_16x16x32_f16(false, a, false, bf, (short)0,
                                                      acc[j], false, false);
    }
    __syncthreads();
  }
  float* yb = Y + (size_t)b * M * Np;
  const float* rb = R ? R + (size_t)b * M * Np : nullptr;
#pragma unroll
  for (int j = 0; j < 4; ++j) {
#pragma unroll
    for (int r = 0; r < 8; ++r) {
      int row = bm0 + wave * 16 + r + (hh << 3);
      size_t o = (size_t)row * Np + bn0 + j * 16 + li;
      float v = acc[j][r] + bias[row];
      if (mode == MODE_RELU)      v = fmaxf(v, 0.f);
      else if (mode == MODE_RES)  v += rb[o];
      else if (mode == MODE_ACC)  v += yb[o];
      yb[o] = v;
    }
  }
}

// ---------------------------------------------------------------------------
// Farthest point sampling: one block (256 thr) per batch. N<=2048 in LDS.
// ---------------------------------------------------------------------------
__global__ void fps_kernel(const float* __restrict__ pts, int N, int m,
                           int* __restrict__ out) {
  __shared__ float px[2048], py[2048], pz[2048], dist[2048];
  __shared__ float rv[256];
  __shared__ int   ri[256];
  __shared__ int   s_last;
  int b = blockIdx.x, tid = threadIdx.x;
  const float* p = pts + (size_t)b * 3 * N;
  for (int t = tid; t < N; t += 256) {
    px[t] = p[t]; py[t] = p[N + t]; pz[t] = p[2 * N + t];
    dist[t] = 1e10f;
  }
  if (tid == 0) { out[(size_t)b * m] = 0; s_last = 0; }
  __syncthreads();
  for (int it = 1; it < m; ++it) {
    int last = s_last;
    float lx = px[last], ly = py[last], lz = pz[last];
    float bv = -1.f; int bidx = 0;
    for (int t = tid; t < N; t += 256) {
      float dx = px[t] - lx, dy = py[t] - ly, dz = pz[t] - lz;
      float d = dx * dx + dy * dy + dz * dz;
      float nd = fminf(dist[t], d);
      dist[t] = nd;
      if (nd > bv) { bv = nd; bidx = t; }
    }
    rv[tid] = bv; ri[tid] = bidx;
    __syncthreads();
    for (int o = 128; o > 0; o >>= 1) {
      if (tid < o) {
        float v2 = rv[tid + o]; int i2 = ri[tid + o];
        if (v2 > rv[tid] || (v2 == rv[tid] && i2 < ri[tid])) { rv[tid] = v2; ri[tid] = i2; }
      }
      __syncthreads();
    }
    if (tid == 0) { s_last = ri[0]; out[(size_t)b * m + it] = ri[0]; }
    __syncthreads();
  }
}

// ---------------------------------------------------------------------------
// gather: out[b,c,m] = in[b,c,idx[b,m]]
// ---------------------------------------------------------------------------
__global__ void gather_kernel(const float* __restrict__ in, const int* __restrict__ idx,
                              int ldIdx, float* __restrict__ out, int C, int Nin, int Mout) {
  int g = blockIdx.x * blockDim.x + threadIdx.x;
  int total = BATCH * C * Mout;
  if (g >= total) return;
  int m = g % Mout, c = (g / Mout) % C, b = g / (Mout * C);
  int id = idx[(size_t)b * ldIdx + m];
  out[g] = in[((size_t)(b * C + c)) * Nin + id];
}

// ---------------------------------------------------------------------------
// KNN: k smallest sq-distances (stable tie-break = lower index, like top_k).
// One thread per query, source points tiled through LDS. Ns % 256 == 0.
// ---------------------------------------------------------------------------
__global__ void knn_kernel(const float* __restrict__ posq, const float* __restrict__ poss,
                           int Nq, int Ns, int k, int* __restrict__ outIdx) {
  __shared__ float sx[256], sy[256], sz[256];
  int b = blockIdx.y;
  int n = blockIdx.x * 256 + threadIdx.x;
  const float* pqb = posq + (size_t)b * 3 * Nq;
  const float* psb = poss + (size_t)b * 3 * Ns;
  float qx = pqb[n], qy = pqb[Nq + n], qz = pqb[2 * Nq + n];
  float bd[16]; int bi[16];
#pragma unroll
  for (int j = 0; j < 16; ++j) { bd[j] = 3.0e38f; bi[j] = 0; }
  for (int t0 = 0; t0 < Ns; t0 += 256) {
    int s = t0 + threadIdx.x;
    sx[threadIdx.x] = psb[s];
    sy[threadIdx.x] = psb[Ns + s];
    sz[threadIdx.x] = psb[2 * Ns + s];
    __syncthreads();
    for (int s2 = 0; s2 < 256; ++s2) {
      float dx = qx - sx[s2], dy = qy - sy[s2], dz = qz - sz[s2];
      float cd = dx * dx + dy * dy + dz * dz;
      int ci = t0 + s2;
#pragma unroll
      for (int j = 0; j < 16; ++j) {
        if (j < k) {
          bool sw = cd < bd[j];
          float td = bd[j]; int ti = bi[j];
          bd[j] = sw ? cd : td; bi[j] = sw ? ci : ti;
          cd = sw ? td : cd;    ci = sw ? ti : ci;
        }
      }
    }
    __syncthreads();
  }
  for (int j = 0; j < k; ++j)
    outIdx[((size_t)b * Nq + n) * k + j] = bi[j];
}

// ---------------------------------------------------------------------------
// 3-NN with inverse-distance weights for three_inter. Ms % 256 == 0.
// ---------------------------------------------------------------------------
__global__ void nn3_kernel(const float* __restrict__ ptgt, int Mt,
                           const float* __restrict__ psrc, int Ms,
                           int* __restrict__ oi, float* __restrict__ ow) {
  __shared__ float sx[256], sy[256], sz[256];
  int b = blockIdx.y;
  int n = blockIdx.x * 256 + threadIdx.x;
  const float* pt = ptgt + (size_t)b * 3 * Mt;
  const float* ps = psrc + (size_t)b * 3 * Ms;
  float qx = pt[n], qy = pt[Mt + n], qz = pt[2 * Mt + n];
  float bd[3] = {3.0e38f, 3.0e38f, 3.0e38f};
  int bi[3] = {0, 0, 0};
  for (int t0 = 0; t0 < Ms; t0 += 256) {
    int s = t0 + threadIdx.x;
    sx[threadIdx.x] = ps[s];
    sy[threadIdx.x] = ps[Ms + s];
    sz[threadIdx.x] = ps[2 * Ms + s];
    __syncthreads();
    for (int s2 = 0; s2 < 256; ++s2) {
      float dx = qx - sx[s2], dy = qy - sy[s2], dz = qz - sz[s2];
      float cd = dx * dx + dy * dy + dz * dz;
      int ci = t0 + s2;
#pragma unroll
      for (int j = 0; j < 3; ++j) {
        bool sw = cd < bd[j];
        float td = bd[j]; int ti = bi[j];
        bd[j] = sw ? cd : td; bi[j] = sw ? ci : ti;
        cd = sw ? td : cd;    ci = sw ? ti : ci;
      }
    }
    __syncthreads();
  }
  float w0 = 1.f / (bd[0] + 1e-8f), w1 = 1.f / (bd[1] + 1e-8f), w2 = 1.f / (bd[2] + 1e-8f);
  float inv = 1.f / (w0 + w1 + w2);
  size_t o = ((size_t)b * Mt + n) * 3;
  oi[o] = bi[0]; oi[o + 1] = bi[1]; oi[o + 2] = bi[2];
  ow[o] = w0 * inv; ow[o + 1] = w1 * inv; ow[o + 2] = w2 * inv;
}

__global__ void interp_kernel(const float* __restrict__ f, const int* __restrict__ oi,
                              const float* __restrict__ ow, float* __restrict__ out,
                              int C, int M1, int M2) {
  int g = blockIdx.x * blockDim.x + threadIdx.x;
  int total = BATCH * C * M2;
  if (g >= total) return;
  int m = g % M2, c = (g / M2) % C, b = g / (M2 * C);
  const int*   ip = oi + ((size_t)b * M2 + m) * 3;
  const float* wp = ow + ((size_t)b * M2 + m) * 3;
  const float* fb = f + ((size_t)(b * C + c)) * M1;
  out[g] = wp[0] * fb[ip[0]] + wp[1] * fb[ip[1]] + wp[2] * fb[ip[2]];
}

// ---------------------------------------------------------------------------
// Fused vector-attention core: one block (4 waves) per query point.
// All WMMA B-operands live in LDS as transposed packed f16 ([col][k], padded
// strides -> contiguous b128 reads, conflict-free banks). f32 kept only for
// scalar consumers (pe for aggregation, logits for softmax).
// ---------------------------------------------------------------------------
struct AttnW {
  const _Float16 *p2w, *a1w, *a2w;
  const float *p1w, *p1b, *pbg, *pbb, *p2b;
  const float *a1b, *abg, *abb, *a2b;
};

__global__ void __launch_bounds__(128)
attn_fused(const float* __restrict__ qB, const float* __restrict__ kB,
           const float* __restrict__ vB, const float* __restrict__ posq,
           const float* __restrict__ poss, const int* __restrict__ knn,
           AttnW w, float* __restrict__ agg, int M, int k) {
  __shared__ float s_pe[64 * 16];                    // pe, f32 [c][j]
  __shared__ float s_lg[64 * 16];                    // logits, f32 [c][j]
  __shared__ __align__(16) _Float16 s_bA[16 * 72];   // B op (pe_pre/attn_in) [j][c pad]
  __shared__ __align__(16) _Float16 s_bH[16 * 264];  // B op H [j][row pad]
  __shared__ float s_pr[3 * 16];
  __shared__ float s_q[64];
  __shared__ int   s_idx[16];

  int n = blockIdx.x, b = blockIdx.y;
  int tid = threadIdx.x;
  int wave = tid >> 5, lane = tid & 31, li = lane & 15, hh = lane >> 4;
  const float* pqb = posq + (size_t)b * 3 * M;
  const float* psb = poss + (size_t)b * 3 * M;

  if (tid < 16) {
    int jj = (tid < k) ? tid : (k - 1);
    s_idx[tid] = knn[((size_t)b * M + n) * k + jj];
  }
  if (tid >= 64 && tid < 128)
    s_q[tid - 64] = qB[((size_t)b * 64 + (tid - 64)) * M + n];
  __syncthreads();
  if (tid < 48) {
    int c = tid >> 4, j = tid & 15;
    s_pr[tid] = pqb[c * M + n] - psb[c * M + s_idx[j]];
  }
  __syncthreads();

  // pe_pre = relu(bn(pos1 @ pos_rel + b)); channel pairs -> packed f16 dwords
  for (int e = tid; e < 16 * 32; e += 128) {
    int j = e >> 5, cp = e & 31;
    int c0 = cp << 1, c1 = c0 + 1;
    float r0 = s_pr[j], r1 = s_pr[16 + j], r2 = s_pr[32 + j];
    float v0 = w.p1b[c0] + w.p1w[c0 * 3] * r0 + w.p1w[c0 * 3 + 1] * r1 + w.p1w[c0 * 3 + 2] * r2;
    float v1 = w.p1b[c1] + w.p1w[c1 * 3] * r0 + w.p1w[c1 * 3 + 1] * r1 + w.p1w[c1 * 3 + 2] * r2;
    v0 = fmaxf(w.pbg[c0] * (v0 * BN_INV) + w.pbb[c0], 0.f);
    v1 = fmaxf(w.pbg[c1] * (v1 * BN_INV) + w.pbb[c1], 0.f);
    H2Pack pk; pk.h[0] = (_Float16)v0; pk.h[1] = (_Float16)v1;
    ((unsigned*)s_bA)[j * 36 + cp] = pk.u;
  }
  __syncthreads();

  // pe = pos2(64x64) @ pe_pre + b  (WMMA; one 16-row tile per wave) -> s_pe f32
  {
    int m0 = wave << 4;
    v8f acc = {};
    for (int kc = 0; kc < 64; kc += 32) {
      v16h a = load_a_frag_h(w.p2w, 64, m0, kc, li, hh);
      v16h bf = load_b_lds_h(s_bA, 72, kc, li, hh);
      acc = __builtin_amdgcn_wmma_f32_16x16x32_f16(false, a, false, bf, (short)0,
                                                   acc, false, false);
    }
    __syncthreads();
#pragma unroll
    for (int r = 0; r < 8; ++r) {
      int row = m0 + r + (hh << 3);
      s_pe[row * 16 + li] = acc[r] + w.p2b[row];
    }
  }
  __syncthreads();

  // attn_in = q - k_gathered + pe  -> overwrite s_bA (f16 transposed)
  for (int e = tid; e < 16 * 32; e += 128) {
    int j = e >> 5, cp = e & 31;
    int c0 = cp << 1, c1 = c0 + 1;
    int id = s_idx[j];
    float v0 = s_q[c0] - kB[((size_t)b * 64 + c0) * M + id] + s_pe[c0 * 16 + j];
    float v1 = s_q[c1] - kB[((size_t)b * 64 + c1) * M + id] + s_pe[c1 * 16 + j];
    H2Pack pk; pk.h[0] = (_Float16)v0; pk.h[1] = (_Float16)v1;
    ((unsigned*)s_bA)[j * 36 + cp] = pk.u;
  }
  __syncthreads();

  // H = relu(bn(attn1(256x64) @ attn_in + b))  (4 tiles/wave) -> s_bH f16
  for (int t = wave; t < 16; t += 4) {
    int m0 = t << 4;
    v8f acc = {};
    for (int kc = 0; kc < 64; kc += 32) {
      v16h a = load_a_frag_h(w.a1w, 64, m0, kc, li, hh);
      v16h bf = load_b_lds_h(s_bA, 72, kc, li, hh);
      acc = __builtin_amdgcn_wmma_f32_16x16x32_f16(false, a, false, bf, (short)0,
                                                   acc, false, false);
    }
#pragma unroll
    for (int r = 0; r < 8; r += 2) {
      int row = m0 + (hh << 3) + r;
      float v0 = acc[r] + w.a1b[row];
      float v1 = acc[r + 1] + w.a1b[row + 1];
      v0 = fmaxf(w.abg[row] * (v0 * BN_INV) + w.abb[row], 0.f);
      v1 = fmaxf(w.abg[row + 1] * (v1 * BN_INV) + w.abb[row + 1], 0.f);
      H2Pack pk; pk.h[0] = (_Float16)v0; pk.h[1] = (_Float16)v1;
      ((unsigned*)s_bH)[li * 132 + (row >> 1)] = pk.u;
    }
  }
  __syncthreads();

  // logits = attn2(64x256) @ H + b  (one tile per wave) -> s_lg f32
  {
    int m0 = wave << 4;
    v8f acc = {};
    for (int kc = 0; kc < 256; kc += 32) {
      v16h a = load_a_frag_h(w.a2w, 256, m0, kc, li, hh);
      v16h bf = load_b_lds_h(s_bH, 264, kc, li, hh);
      acc = __builtin_amdgcn_wmma_f32_16x16x32_f16(false, a, false, bf, (short)0,
                                                   acc, false, false);
    }
#pragma unroll
    for (int r = 0; r < 8; ++r) {
      int row = m0 + r + (hh << 3);
      s_lg[row * 16 + li] = acc[r] + w.a2b[row];
    }
  }
  __syncthreads();

  // per-channel softmax over the k valid neighbors + aggregation
  if (tid < 64) {
    int c = tid;
    float mx = -3.0e38f;
    for (int j = 0; j < k; ++j) mx = fmaxf(mx, s_lg[c * 16 + j]);
    float sum = 0.f;
    float ex[16];
    for (int j = 0; j < k; ++j) { float e = expf(s_lg[c * 16 + j] - mx); ex[j] = e; sum += e; }
    float inv = 1.f / sum;
    float a = 0.f;
    for (int j = 0; j < k; ++j) {
      float vg = vB[((size_t)b * 64 + c) * M + s_idx[j]] + s_pe[c * 16 + j];
      a += ex[j] * inv * vg;
    }
    agg[((size_t)b * 64 + c) * M + n] = a;
  }
}

// ---------------------------------------------------------------------------
// Host-side orchestration
// ---------------------------------------------------------------------------
extern "C" void kernel_launch(void* const* d_in, const int* in_sizes, int n_in,
                              void* d_out, int out_size, void* d_ws, size_t ws_size,
                              hipStream_t stream) {
  (void)in_sizes; (void)n_in; (void)out_size; (void)ws_size;
  const float* pq = (const float*)d_in[0];
  const float* fq = (const float*)d_in[1];
  const float* ps = (const float*)d_in[2];
  const float* fs = (const float*)d_in[3];
  // attn[i]: q.w q.b k.w k.b v.w v.b pos1.w pos1.b pbn.g pbn.bt pos2.w pos2.b
  //          attn1.w attn1.b abn.g abn.bt attn2.w attn2.b end.w end.b
  const float* ap[3][20];
  for (int i = 0; i < 3; ++i)
    for (int j = 0; j < 20; ++j) ap[i][j] = (const float*)d_in[4 + i * 20 + j];
  const float* qm[2][6];
  const float* sm[2][6];
  for (int i = 0; i < 2; ++i)
    for (int j = 0; j < 6; ++j) qm[i][j] = (const float*)d_in[64 + i * 6 + j];
  for (int i = 0; i < 2; ++i)
    for (int j = 0; j < 6; ++j) sm[i][j] = (const float*)d_in[76 + i * 6 + j];

  // workspace arena (~68 MB)
  char* wsp = (char*)d_ws;
  size_t off = 0;
  auto alc = [&](size_t bytes) -> void* {
    void* p = wsp + off;
    off += (bytes + 255) & ~(size_t)255;
    return p;
  };
  int*   fpsq  = (int*)alc(8 * 512 * 4);
  int*   fpss  = (int*)alc(8 * 512 * 4);
  float* posqA = (float*)alc((size_t)8 * 3 * 256 * 4);
  float* posqB = (float*)alc((size_t)8 * 3 * 512 * 4);
  float* possL = (float*)alc((size_t)8 * 3 * 512 * 4);
  float* fg1   = (float*)alc((size_t)8 * 128 * 512 * 4);
  float* fg2   = (float*)alc((size_t)8 * 128 * 512 * 4);
  float* proj1 = (float*)alc((size_t)8 * 128 * 2048 * 4);
  float* proj2 = (float*)alc((size_t)8 * 128 * 2048 * 4);
  float* f1c   = (float*)alc((size_t)8 * 128 * 2048 * 4);
  float* f2c   = (float*)alc((size_t)8 * 128 * 2048 * 4);
  float* tmp   = (float*)alc((size_t)8 * 128 * 2048 * 4);
  float* fbA   = (float*)alc((size_t)8 * 128 * 256 * 4);
  float* fbB   = (float*)alc((size_t)8 * 128 * 512 * 4);
  float* qb    = (float*)alc((size_t)8 * 64 * 2048 * 4);
  float* kb    = (float*)alc((size_t)8 * 64 * 2048 * 4);
  float* vb    = (float*)alc((size_t)8 * 64 * 2048 * 4);
  float* agg   = (float*)alc((size_t)8 * 64 * 2048 * 4);
  int*   knnI  = (int*)alc((size_t)8 * 2048 * 16 * 4);
  int*   nn3i  = (int*)alc((size_t)8 * 2048 * 3 * 4);
  float* nn3w  = (float*)alc((size_t)8 * 2048 * 3 * 4);
  // f16 attention weights: per level pos2 (64x64), attn1 (256x64), attn2 (64x256)
  _Float16* hP2[3]; _Float16* hA1[3]; _Float16* hA2[3];
  for (int i = 0; i < 3; ++i) {
    hP2[i] = (_Float16*)alc(4096 * 2);
    hA1[i] = (_Float16*)alc(16384 * 2);
    hA2[i] = (_Float16*)alc(16384 * 2);
  }

  auto cvt = [&](const float* src, _Float16* dst, int n) {
    cvt_f16_kernel<<<dim3((n + 255) / 256), dim3(256), 0, stream>>>(src, dst, n);
  };
  for (int i = 0; i < 3; ++i) {
    cvt(ap[i][10], hP2[i], 4096);
    cvt(ap[i][12], hA1[i], 16384);
    cvt(ap[i][16], hA2[i], 16384);
  }

  auto gemm = [&](const float* W, const float* bias, const float* X1, const float* X2,
                  int C1, float* Y, const float* R, int M, int K, int Np, int mode) {
    gemm_conv<<<dim3(Np / 64, M / 64, 8), dim3(128), 0, stream>>>(W, bias, X1, X2, C1,
                                                                  Y, R, M, K, Np, mode);
  };
  auto gath = [&](const float* in, const int* idx, float* out, int C, int Mout) {
    int total = 8 * C * Mout;
    gather_kernel<<<dim3((total + 255) / 256), dim3(256), 0, stream>>>(in, idx, 512, out,
                                                                       C, 2048, Mout);
  };
  auto interp = [&](const float* f, float* out, int M1, int M2) {
    int total = 8 * 128 * M2;
    interp_kernel<<<dim3((total + 255) / 256), dim3(256), 0, stream>>>(f, nn3i, nn3w, out,
                                                                       128, M1, M2);
  };
  auto mlp = [&](const float* const* m, const float* xa, const float* xb, float* t,
                 float* out, int Np) {
    gemm(m[4], m[5], xa, xb, 128, out, nullptr, 128, 256, Np, MODE_BIAS);    // shortcut
    gemm(m[0], m[1], xa, xb, 128, t, nullptr, 128, 256, Np, MODE_RELU);      // c1 + relu
    gemm(m[2], m[3], t, nullptr, 128, out, nullptr, 128, 128, Np, MODE_ACC); // c2 +=
  };
  auto attention = [&](int lvl, const float* pQ, const float* f1, const float* pS,
                       const float* f2, int M, int kk, float* fout) {
    const float* const* a = ap[lvl];
    gemm(a[0], a[1], f1, nullptr, 128, qb, nullptr, 64, 128, M, MODE_BIAS);
    gemm(a[2], a[3], f2, nullptr, 128, kb, nullptr, 64, 128, M, MODE_BIAS);
    gemm(a[4], a[5], f2, nullptr, 128, vb, nullptr, 64, 128, M, MODE_BIAS);
    knn_kernel<<<dim3(M / 256, 8), dim3(256), 0, stream>>>(pQ, pS, M, M, kk, knnI);
    AttnW w = {hP2[lvl], hA1[lvl], hA2[lvl],
               a[6], a[7], a[8], a[9], a[11], a[13], a[14], a[15], a[17]};
    attn_fused<<<dim3(M, 8), dim3(128), 0, stream>>>(qb, kb, vb, pQ, pS, knnI, w, agg, M, kk);
    gemm(a[18], a[19], agg, nullptr, 64, fout, f1, 128, 64, M, MODE_RES);  // end + residual
  };

  // ---- FPS (prefix property: 256-sequence is first 256 of 512-sequence) ----
  fps_kernel<<<dim3(8), dim3(256), 0, stream>>>(pq, 2048, 512, fpsq);
  fps_kernel<<<dim3(8), dim3(256), 0, stream>>>(ps, 2048, 512, fpss);

  // ---- level 2 (M=256, knn=8) ----
  gath(pq, fpsq, posqA, 3, 256);
  gath(ps, fpss, possL, 3, 256);
  gath(fq, fpsq, fg1, 128, 256);
  gath(fs, fpss, fg2, 128, 256);
  attention(2, posqA, fg1, possL, fg2, 256, 8, fbA);

  // ---- level 1 (M=512, knn=12) ----
  gath(pq, fpsq, posqB, 3, 512);
  gath(ps, fpss, possL, 3, 512);
  gath(fq, fpsq, fg1, 128, 512);
  gath(fs, fpss, fg2, 128, 512);
  nn3_kernel<<<dim3(512 / 256, 8), dim3(256), 0, stream>>>(posqB, 512, posqA, 256, nn3i, nn3w);
  interp(fbA, proj1, 256, 512);
  nn3_kernel<<<dim3(512 / 256, 8), dim3(256), 0, stream>>>(possL, 512, posqA, 256, nn3i, nn3w);
  interp(fbA, proj2, 256, 512);
  mlp(qm[1], fg1, proj1, tmp, f1c, 512);
  mlp(sm[1], fg2, proj2, tmp, f2c, 512);
  attention(1, posqB, f1c, possL, f2c, 512, 12, fbB);

  // ---- level 0 (M=2048, knn=16) ----
  nn3_kernel<<<dim3(2048 / 256, 8), dim3(256), 0, stream>>>(pq, 2048, posqB, 512, nn3i, nn3w);
  interp(fbB, proj1, 512, 2048);
  nn3_kernel<<<dim3(2048 / 256, 8), dim3(256), 0, stream>>>(ps, 2048, posqB, 512, nn3i, nn3w);
  interp(fbB, proj2, 512, 2048);
  mlp(qm[0], fq, proj1, tmp, f1c, 2048);
  mlp(sm[0], fs, proj2, tmp, f2c, 2048);
  attention(0, pq, f1c, ps, f2c, 2048, 16, (float*)d_out);
}
